// DecodingLayer_44032004719074
// MI455X (gfx1250) — matrix operand order repaired and checked by reading, not compiled, not run on hardware
//
#include <hip/hip_runtime.h>
#include <math.h>

// ---------------------------------------------------------------------------
// Problem constants (from reference setup_inputs)
// ---------------------------------------------------------------------------
constexpr int BB = 2;
constexpr int SS = 2048;
constexpr int DD = 1024;
constexpr int HH = 16;
constexpr int DH = 64;      // DD / HH
constexpr int FF = 1024;
constexpr int MM = BB * SS; // 4096 rows of activations

typedef __attribute__((ext_vector_type(16))) _Float16 v16h;
typedef __attribute__((ext_vector_type(8)))  _Float16 v8h;
typedef __attribute__((ext_vector_type(8)))  float    v8f;

// ---------------------------------------------------------------------------
// WMMA fragment loaders (wave32, v_wmma_f32_16x16x32_f16)
//
// A-matrix 16x32 f16 (ISA 7.12.2): lane m = lane&15, hi = lane>>4
//   elements 0..7  = K (hi*8)   .. (hi*8)+7     of row m
//   elements 8..15 = K (hi*8)+16.. (hi*8)+23    of row m
// B-matrix 32x16 f16 (same K order as A): lane n = lane&15
//   elements 0..7  = K (hi*8)   .. (hi*8)+7     of column n
//   elements 8..15 = K (hi*8)+16.. (hi*8)+23    of column n
// Both loaders take a row-major tile where the 16-wide dim (M or N) is rows
// and K runs along columns.  src must be 16B aligned, ld a multiple of 8.
// ---------------------------------------------------------------------------
__device__ __forceinline__ v16h load_a_frag(const _Float16* base, int ld) {
    int lane = threadIdx.x & 31;
    int m = lane & 15, hi = lane >> 4;
    const _Float16* p = base + (size_t)m * ld + hi * 8;
    v8h lo = *(const v8h*)(p);
    v8h hc = *(const v8h*)(p + 16);
    v16h r;
#pragma unroll
    for (int i = 0; i < 8; ++i) { r[i] = lo[i]; r[8 + i] = hc[i]; }
    return r;
}

__device__ __forceinline__ v16h load_b_frag(const _Float16* base, int ld) {
    int lane = threadIdx.x & 31;
    int n = lane & 15, hi = lane >> 4;
    const _Float16* p = base + (size_t)n * ld + hi * 8;
    v8h x0 = *(const v8h*)(p);
    v8h x1 = *(const v8h*)(p + 16);
    v16h r;
#pragma unroll
    for (int i = 0; i < 8; ++i) { r[i] = x0[i]; r[8 + i] = x1[i]; }
    return r;
}

__device__ __forceinline__ v8f wmma_f16(v16h a, v16h b, v8f c) {
    return __builtin_amdgcn_wmma_f32_16x16x32_f16(
        /*neg_a=*/false, a, /*neg_b=*/false, b,
        /*c_mod=*/(short)0, c, /*reuse_a=*/false, /*reuse_b=*/false);
}

// ---------------------------------------------------------------------------
// Elementwise fp32 -> f16 conversion
// ---------------------------------------------------------------------------
__global__ void cvt_f16_kernel(const float* __restrict__ in,
                               _Float16* __restrict__ out, size_t n) {
    size_t i = (size_t)blockIdx.x * blockDim.x + threadIdx.x;
    size_t stride = (size_t)gridDim.x * blockDim.x;
    for (; i < n; i += stride) out[i] = (_Float16)in[i];
}

// Weight transpose + convert:  Wt[n*K + k] = (f16) W[k*N + n]
__global__ void transw_kernel(const float* __restrict__ W,
                              _Float16* __restrict__ Wt, int K, int N) {
    size_t total = (size_t)K * N;
    size_t i = (size_t)blockIdx.x * blockDim.x + threadIdx.x;
    size_t stride = (size_t)gridDim.x * blockDim.x;
    for (; i < total; i += stride) {
        int nn = (int)(i / K);
        int kk = (int)(i % K);
        Wt[i] = (_Float16)W[(size_t)kk * N + nn];
    }
}

// V transpose per head:  Vt[((b*H+h)*DH + d)*S + s] = V[(b*S+s)*D + h*DH + d]
__global__ void transv_kernel(const _Float16* __restrict__ V,
                              _Float16* __restrict__ Vt) {
    size_t total = (size_t)BB * HH * DH * SS;
    size_t i = (size_t)blockIdx.x * blockDim.x + threadIdx.x;
    size_t stride = (size_t)gridDim.x * blockDim.x;
    for (; i < total; i += stride) {
        int s = (int)(i % SS);
        size_t r = i / SS;
        int d = (int)(r % DH); r /= DH;
        int h = (int)(r % HH);
        int b = (int)(r / HH);
        Vt[i] = V[((size_t)b * SS + s) * DD + h * DH + d];
    }
}

// ---------------------------------------------------------------------------
// GEMM: C[M,N] = A[M,K] @ Bt[N,K]^T + bias, optional ReLU,
// optional fp32 and/or f16 outputs.  One wave owns a 32x32 tile.
// ---------------------------------------------------------------------------
__global__ void __launch_bounds__(128)
gemm_kernel(const _Float16* __restrict__ A, const _Float16* __restrict__ Bt,
            const float* __restrict__ bias,
            float* __restrict__ Cf, _Float16* __restrict__ Ch,
            int M, int N, int K, int relu) {
    int wave = threadIdx.x >> 5;
    int tile = blockIdx.x * 4 + wave;
    int ntn = N >> 5;
    int tm = tile / ntn;
    int tn = tile % ntn;
    if (tm >= (M >> 5)) return;

    v8f acc[2][2];
#pragma unroll
    for (int i = 0; i < 2; ++i)
#pragma unroll
        for (int j = 0; j < 2; ++j)
            acc[i][j] = v8f{0.f, 0.f, 0.f, 0.f, 0.f, 0.f, 0.f, 0.f};

    const _Float16* Abase = A + (size_t)(tm * 32) * K;
    const _Float16* Bbase = Bt + (size_t)(tn * 32) * K;

    for (int kc = 0; kc < K; kc += 32) {
        v16h a0 = load_a_frag(Abase + kc, K);
        v16h a1 = load_a_frag(Abase + (size_t)16 * K + kc, K);
        v16h b0 = load_b_frag(Bbase + kc, K);
        v16h b1 = load_b_frag(Bbase + (size_t)16 * K + kc, K);
        acc[0][0] = wmma_f16(a0, b0, acc[0][0]);
        acc[0][1] = wmma_f16(a0, b1, acc[0][1]);
        acc[1][0] = wmma_f16(a1, b0, acc[1][0]);
        acc[1][1] = wmma_f16(a1, b1, acc[1][1]);
    }

    int lane = threadIdx.x & 31;
    int n0 = lane & 15, hi = lane >> 4;
#pragma unroll
    for (int i = 0; i < 2; ++i) {
#pragma unroll
        for (int j = 0; j < 2; ++j) {
            int col = tn * 32 + j * 16 + n0;
            float bv = bias ? bias[col] : 0.f;
#pragma unroll
            for (int r = 0; r < 8; ++r) {
                int row = tm * 32 + i * 16 + hi * 8 + r;
                float v = acc[i][j][r] + bv;
                if (relu) v = fmaxf(v, 0.f);
                size_t o = (size_t)row * N + col;
                if (Cf) Cf[o] = v;
                if (Ch) Ch[o] = (_Float16)v;
            }
        }
    }
}

// ---------------------------------------------------------------------------
// Attention: one wave per (b, h, q-tile of 16 rows).
// Phase 1: scores(16 x S) via WMMA -> LDS (fp32, 128KB dynamic).
// Phase 2: exact row softmax with pre-scale and post-softmax triu mask
//          (reference quirks: scale by 1/sqrt(S); mask keeps k >= q AFTER
//           normalization, so the denominator spans all k).
// Phase 3: P @ V via WMMA (P fragments converted f32->f16 from LDS,
//          V from pre-transposed [B,H,DH,S] buffer).
// Z written in [B,H,S,DH] contiguous order == the buggy merge layout.
// ---------------------------------------------------------------------------
__global__ void __launch_bounds__(32)
attn_kernel(const _Float16* __restrict__ Q, const _Float16* __restrict__ K,
            const _Float16* __restrict__ Vt, float* __restrict__ Z,
            float scale, int mask) {
    extern __shared__ float smem[]; // 16 * SS floats
    int qt = blockIdx.x & ((SS / 16) - 1);
    int bh = blockIdx.x >> 7;       // SS/16 = 128
    int h = bh & (HH - 1);
    int b = bh / HH;

    int lane = threadIdx.x & 31;
    int n0 = lane & 15, hi = lane >> 4;

    const _Float16* Qb = Q + (size_t)b * SS * DD + (size_t)(qt * 16) * DD + h * DH;
    v16h aq0 = load_a_frag(Qb, DD);
    v16h aq1 = load_a_frag(Qb + 32, DD);

    // ---- phase 1: QK^T score tiles into LDS
    const _Float16* Kb = K + (size_t)b * SS * DD + h * DH;
    for (int kt = 0; kt < SS / 16; ++kt) {
        const _Float16* Kt = Kb + (size_t)(kt * 16) * DD;
        v16h bk0 = load_b_frag(Kt, DD);
        v16h bk1 = load_b_frag(Kt + 32, DD);
        v8f c = v8f{0.f, 0.f, 0.f, 0.f, 0.f, 0.f, 0.f, 0.f};
        c = wmma_f16(aq0, bk0, c);
        c = wmma_f16(aq1, bk1, c);
#pragma unroll
        for (int r = 0; r < 8; ++r)
            smem[(size_t)(hi * 8 + r) * SS + kt * 16 + n0] = c[r];
    }
    __syncthreads();

    // ---- phase 2: softmax per row (all 32 lanes cooperate on one row)
    for (int m = 0; m < 16; ++m) {
        int qg = qt * 16 + m;
        float mx = -3.402823466e38f;
        for (int c = lane; c < SS; c += 32)
            mx = fmaxf(mx, smem[(size_t)m * SS + c] * scale);
#pragma unroll
        for (int o = 16; o > 0; o >>= 1) mx = fmaxf(mx, __shfl_xor(mx, o, 32));
        float sum = 0.f;
        for (int c = lane; c < SS; c += 32)
            sum += expf(smem[(size_t)m * SS + c] * scale - mx);
#pragma unroll
        for (int o = 16; o > 0; o >>= 1) sum += __shfl_xor(sum, o, 32);
        float inv = 1.f / sum;
        for (int c = lane; c < SS; c += 32) {
            float p = expf(smem[(size_t)m * SS + c] * scale - mx) * inv;
            if (mask && c < qg) p = 0.f;   // triu keeps k >= q
            smem[(size_t)m * SS + c] = p;
        }
    }
    __syncthreads();

    // ---- phase 3: Z = P @ V
    v8f acc[4];
#pragma unroll
    for (int nt = 0; nt < 4; ++nt)
        acc[nt] = v8f{0.f, 0.f, 0.f, 0.f, 0.f, 0.f, 0.f, 0.f};

    const _Float16* Vb = Vt + ((size_t)(b * HH + h) * DH) * SS;
    for (int kc = 0; kc < SS; kc += 32) {
        // P fragment (A layout) from LDS fp32, converted to f16
        const float* prow = smem + (size_t)(lane & 15) * SS + kc + hi * 8;
        float4 f0 = *(const float4*)(prow);
        float4 f1 = *(const float4*)(prow + 4);
        float4 f2 = *(const float4*)(prow + 16);
        float4 f3 = *(const float4*)(prow + 20);
        v16h ap;
        ap[0] = (_Float16)f0.x;  ap[1] = (_Float16)f0.y;
        ap[2] = (_Float16)f0.z;  ap[3] = (_Float16)f0.w;
        ap[4] = (_Float16)f1.x;  ap[5] = (_Float16)f1.y;
        ap[6] = (_Float16)f1.z;  ap[7] = (_Float16)f1.w;
        ap[8] = (_Float16)f2.x;  ap[9] = (_Float16)f2.y;
        ap[10] = (_Float16)f2.z; ap[11] = (_Float16)f2.w;
        ap[12] = (_Float16)f3.x; ap[13] = (_Float16)f3.y;
        ap[14] = (_Float16)f3.z; ap[15] = (_Float16)f3.w;
#pragma unroll
        for (int nt = 0; nt < 4; ++nt) {
            v16h bv = load_b_frag(Vb + (size_t)(nt * 16) * SS + kc, SS);
            acc[nt] = wmma_f16(ap, bv, acc[nt]);
        }
    }

    // Z layout: [B][H][S][DH] contiguous (== buggy merge when read as [B,S,D])
    float* Zb = Z + (size_t)b * SS * DD + (size_t)h * SS * DH + (size_t)(qt * 16) * DH;
#pragma unroll
    for (int nt = 0; nt < 4; ++nt)
#pragma unroll
        for (int r = 0; r < 8; ++r)
            Zb[(size_t)(hi * 8 + r) * DH + nt * 16 + n0] = acc[nt][r];
}

// ---------------------------------------------------------------------------
// out = LayerNorm(resid + z) * g + b ; writes fp32 (+ optional f16 copy)
// One 256-thread block per row of D=1024.
// ---------------------------------------------------------------------------
__global__ void __launch_bounds__(256)
add_ln_kernel(const float* __restrict__ resid, const float* __restrict__ z,
              const float* __restrict__ g, const float* __restrict__ be,
              float* __restrict__ outF, _Float16* __restrict__ outH) {
    __shared__ float red[256];
    size_t row = blockIdx.x;
    const float* r0 = resid + row * DD;
    const float* z0 = z + row * DD;
    int tid = threadIdx.x;

    float t[4];
    float s = 0.f;
#pragma unroll
    for (int i = 0; i < 4; ++i) {
        int c = tid + i * 256;
        t[i] = r0[c] + z0[c];
        s += t[i];
    }
    red[tid] = s;
    __syncthreads();
    for (int st = 128; st > 0; st >>= 1) {
        if (tid < st) red[tid] += red[tid + st];
        __syncthreads();
    }
    float mean = red[0] * (1.f / DD);
    __syncthreads();

    float vs = 0.f;
#pragma unroll
    for (int i = 0; i < 4; ++i) {
        float d = t[i] - mean;
        vs += d * d;
    }
    red[tid] = vs;
    __syncthreads();
    for (int st = 128; st > 0; st >>= 1) {
        if (tid < st) red[tid] += red[tid + st];
        __syncthreads();
    }
    float inv = rsqrtf(red[0] * (1.f / DD) + 1e-5f);

#pragma unroll
    for (int i = 0; i < 4; ++i) {
        int c = tid + i * 256;
        float y = (t[i] - mean) * inv * g[c] + be[c];
        outF[row * DD + c] = y;
        if (outH) outH[row * DD + c] = (_Float16)y;
    }
}

// ---------------------------------------------------------------------------
// Host-side orchestration
// ---------------------------------------------------------------------------
extern "C" void kernel_launch(void* const* d_in, const int* in_sizes, int n_in,
                              void* d_out, int out_size, void* d_ws, size_t ws_size,
                              hipStream_t stream) {
    (void)in_sizes; (void)n_in; (void)out_size; (void)ws_size;

    const float* enc   = (const float*)d_in[0];
    const float* dec   = (const float*)d_in[1];
    const float* sa_w[3] = {(const float*)d_in[2], (const float*)d_in[4], (const float*)d_in[6]};
    const float* sa_bv[3] = {(const float*)d_in[3], (const float*)d_in[5], (const float*)d_in[7]};
    const float* sa_g = (const float*)d_in[8],  *sa_b = (const float*)d_in[9];
    const float* ca_w[3] = {(const float*)d_in[10], (const float*)d_in[12], (const float*)d_in[14]};
    const float* ca_bv[3] = {(const float*)d_in[11], (const float*)d_in[13], (const float*)d_in[15]};
    const float* ca_g = (const float*)d_in[16], *ca_b = (const float*)d_in[17];
    const float* w1 = (const float*)d_in[18], *b1 = (const float*)d_in[19];
    const float* w2 = (const float*)d_in[20], *b2 = (const float*)d_in[21];
    const float* ffn_g = (const float*)d_in[22], *ffn_b = (const float*)d_in[23];
    float* out = (float*)d_out;

    const size_t MD = (size_t)MM * DD;

    // bump allocator over workspace
    char* w = (char*)d_ws;
    auto alloc = [&](size_t bytes) -> void* {
        void* p = (void*)w;
        w += (bytes + 255) & ~(size_t)255;
        return p;
    };
    _Float16* decH = (_Float16*)alloc(MD * 2);   // reused as FFN hidden f16
    _Float16* encH = (_Float16*)alloc(MD * 2);
    _Float16* xH   = (_Float16*)alloc(MD * 2);   // f16 copy of x1 / x2
    _Float16* qH   = (_Float16*)alloc(MD * 2);
    _Float16* kH   = (_Float16*)alloc(MD * 2);
    _Float16* vH   = (_Float16*)alloc(MD * 2);
    _Float16* vtH  = (_Float16*)alloc(MD * 2);
    _Float16* wt[8];
    for (int i = 0; i < 8; ++i) wt[i] = (_Float16*)alloc((size_t)DD * DD * 2);
    float* Zf  = (float*)alloc(MD * 4);          // also FFN y
    float* x1f = (float*)alloc(MD * 4);
    float* x2f = (float*)alloc(MD * 4);

    const int TPB = 256;
    const int cvtBlocks = 4096;
    const float saScale = 1.f / sqrtf((float)SS);

    (void)hipFuncSetAttribute((const void*)attn_kernel,
                              hipFuncAttributeMaxDynamicSharedMemorySize,
                              16 * SS * (int)sizeof(float));

    // activation conversions
    cvt_f16_kernel<<<cvtBlocks, TPB, 0, stream>>>(dec, decH, MD);
    cvt_f16_kernel<<<cvtBlocks, TPB, 0, stream>>>(enc, encH, MD);

    // weight transposes (all [1024,1024])
    const float* wsrc[8] = {sa_w[0], sa_w[1], sa_w[2], ca_w[0], ca_w[1], ca_w[2], w1, w2};
    for (int i = 0; i < 8; ++i)
        transw_kernel<<<2048, TPB, 0, stream>>>(wsrc[i], wt[i], DD, DD);

    const int gemmBlocks = (MM / 32) * (DD / 32) / 4;  // 1024
    const size_t attnLds = (size_t)16 * SS * sizeof(float);

    // ---- self-attention (scale by 1/sqrt(S), post-softmax triu mask)
    gemm_kernel<<<gemmBlocks, 128, 0, stream>>>(decH, wt[0], sa_bv[0], nullptr, qH, MM, DD, DD, 0);
    gemm_kernel<<<gemmBlocks, 128, 0, stream>>>(decH, wt[1], sa_bv[1], nullptr, kH, MM, DD, DD, 0);
    gemm_kernel<<<gemmBlocks, 128, 0, stream>>>(decH, wt[2], sa_bv[2], nullptr, vH, MM, DD, DD, 0);
    transv_kernel<<<cvtBlocks, TPB, 0, stream>>>(vH, vtH);
    attn_kernel<<<BB * HH * (SS / 16), 32, attnLds, stream>>>(qH, kH, vtH, Zf, saScale, 1);
    add_ln_kernel<<<MM, 256, 0, stream>>>(dec, Zf, sa_g, sa_b, x1f, xH);

    // ---- cross-attention (q from x, k/v from encoded; no scale, no mask)
    gemm_kernel<<<gemmBlocks, 128, 0, stream>>>(xH,   wt[3], ca_bv[0], nullptr, qH, MM, DD, DD, 0);
    gemm_kernel<<<gemmBlocks, 128, 0, stream>>>(encH, wt[4], ca_bv[1], nullptr, kH, MM, DD, DD, 0);
    gemm_kernel<<<gemmBlocks, 128, 0, stream>>>(encH, wt[5], ca_bv[2], nullptr, vH, MM, DD, DD, 0);
    transv_kernel<<<cvtBlocks, TPB, 0, stream>>>(vH, vtH);
    attn_kernel<<<BB * HH * (SS / 16), 32, attnLds, stream>>>(qH, kH, vtH, Zf, 1.f, 0);
    add_ln_kernel<<<MM, 256, 0, stream>>>(x1f, Zf, ca_g, ca_b, x2f, xH);

    // ---- FFN: relu(x @ w1 + b1) @ w2 + b2, residual + LN -> d_out
    gemm_kernel<<<gemmBlocks, 128, 0, stream>>>(xH,   wt[6], b1, nullptr, decH, MM, FF, DD, 1);
    gemm_kernel<<<gemmBlocks, 128, 0, stream>>>(decH, wt[7], b2, Zf, nullptr, MM, DD, FF, 0);
    add_ln_kernel<<<MM, 256, 0, stream>>>(x2f, Zf, ffn_g, ffn_b, out, nullptr);
}
